// CMA_34479997453023
// MI455X (gfx1250) — matrix operand order repaired and checked by reading, not compiled, not run on hardware
//
#include <hip/hip_runtime.h>
#include <hip/hip_bf16.h>

typedef __attribute__((ext_vector_type(16))) _Float16       v16h;
typedef __attribute__((ext_vector_type(8)))  float          v8f;
typedef __attribute__((ext_vector_type(8)))  unsigned short u16x8;
typedef __attribute__((ext_vector_type(16))) unsigned short u16x16;

#define NUM_CLASSES 1000
#define CPAD        1024        // classes padded to 64 WMMA tiles of 16
#define FEAT_DIM    2048
#define NSAMP       65536
#define SIGMA       0.2f
#define KSPLIT      4           // sample-range split for parallelism
#define ONE_HOT_INV 16777216.0f // one-hot value is f16 0x0001 = 2^-24

// ---------------------------------------------------------------------------
// Kernel 1: zero the partial-sum buffer (atomic accumulation target)
// ---------------------------------------------------------------------------
__global__ void cma_zero_sums(float* __restrict__ sums) {
    size_t i = (size_t)blockIdx.x * blockDim.x + threadIdx.x;
    if (i < (size_t)2 * CPAD * FEAT_DIM) sums[i] = 0.0f;
}

__global__ void cma_zero_counts(unsigned int* __restrict__ counts) {
    int i = blockIdx.x * blockDim.x + threadIdx.x;
    if (i < 2 * CPAD) counts[i] = 0u;
}

// ---------------------------------------------------------------------------
// Kernel 2: histogram the labels (integer atomics -> exact)
// ---------------------------------------------------------------------------
__global__ void cma_count(const int* __restrict__ l_rgb,
                          const int* __restrict__ l_ir,
                          unsigned int* __restrict__ counts) {
    int i = blockIdx.x * blockDim.x + threadIdx.x;
    if (i < NSAMP) {
        atomicAdd(&counts[l_rgb[i]], 1u);
    } else if (i < 2 * NSAMP) {
        atomicAdd(&counts[CPAD + l_ir[i - NSAMP]], 1u);
    }
}

// ---------------------------------------------------------------------------
// Kernel 3: segment-sum as one-hot GEMM via v_wmma_f32_16x16x32_f16.
//   sums[c, d] = 2^-24 * sum_{i : label_i == c} feats[i, d]
//
// Workgroup = 512 threads (16 waves), owns a 64-wide feature slice, one
// modality, and one of KSPLIT sample ranges. Wave w accumulates class tiles
// [w*4, w*4+4) x 4 feature tiles (16 f32 accumulators = 128 VGPRs), so the
// 16 waves cover all CPAD=1024 classes and one A build feeds 4 WMMAs.
//
// One-hot A build (per class tile, packed u16 pairs, 2 VALU / element pair):
//   d   = label - cls            (mod 2^16; nonzero iff mismatch, wraps big)
//   ind = sat_sub(1, d)          (u16 saturating: 1 iff match else 0)
// 0x0001 reinterpreted as f16 is the subnormal 2^-24 (WMMA preserves denorms)
// -> sums are scaled by 2^-24, compensated exactly in the finalize kernel.
// The A build is software-pipelined (a / a_next) so its pk-sub writes never
// hit the A registers of the WMMA group just issued (kills the 4x v_nop
// WMMA->VALU WAR hazard fill the compiler otherwise inserts).
// ---------------------------------------------------------------------------
__global__ __launch_bounds__(512) void cma_scatter_wmma(
    const float* __restrict__ feats_rgb,
    const float* __restrict__ feats_ir,
    const int*   __restrict__ lab_rgb,
    const int*   __restrict__ lab_ir,
    float*       __restrict__ sums)        // [2][CPAD][FEAT_DIM]
{
    const int mod = blockIdx.z;
    const float* __restrict__ feats  = mod ? feats_ir : feats_rgb;
    const int*   __restrict__ labels = mod ? lab_ir   : lab_rgb;
    float* __restrict__ msums = sums + (size_t)mod * CPAD * FEAT_DIM;

    const int f0    = blockIdx.x * 64;               // feature-slice base
    const int kbase = blockIdx.y * (NSAMP / KSPLIT); // sample-range base
    const int tid   = threadIdx.x;
    const int lane  = tid & 31;           // wave32
    const int wave  = tid >> 5;           // 0..15
    const int n     = lane & 15;          // B/D column within tile
    const int h     = lane >> 4;          // K-half selector

    // f16 feats tile, transposed: tileT[feature][k]. Row stride = 64B, so a
    // lane's 16-half B fragment is one contiguous 32B LDS read.
    __shared__ __align__(64) _Float16 tileT[64][32];
    __shared__ __align__(16) unsigned short labs16[32];

    v8f acc[4][4];
    {
        const v8f vz = {0.f, 0.f, 0.f, 0.f, 0.f, 0.f, 0.f, 0.f};
#pragma unroll
        for (int t = 0; t < 4; ++t)
#pragma unroll
            for (int ft = 0; ft < 4; ++ft) acc[t][ft] = vz;
    }

    u16x16 ones;
#pragma unroll
    for (int e = 0; e < 16; ++e) ones[e] = 1;

    const int srow = tid >> 4;            // staging: sample row 0..31
    const int scol = (tid & 15) * 4;      // staging: feature col 0,4,...,60

    for (int kc = 0; kc < NSAMP / KSPLIT; kc += 32) {
        const int k0 = kbase + kc;
        __syncthreads();   // protect LDS reuse from previous iteration

        // ---- stage: all 512 threads load float4, convert, store transposed
        {
            const float4 v = *(const float4*)(feats +
                                (size_t)(k0 + srow) * FEAT_DIM + f0 + scol);
            tileT[scol + 0][srow] = (_Float16)v.x;
            tileT[scol + 1][srow] = (_Float16)v.y;
            tileT[scol + 2][srow] = (_Float16)v.z;
            tileT[scol + 3][srow] = (_Float16)v.w;
            if (tid < 32) labs16[tid] = (unsigned short)labels[k0 + tid];
            if (kc + 32 < NSAMP / KSPLIT)
                __builtin_prefetch(feats + (size_t)(k0 + 32 + srow) * FEAT_DIM
                                          + f0 + scol, 0, 3);
        }
        __syncthreads();

        // ---- B fragments: 32x16 f16; lane (n + 16h) holds column n,
        //      K = 16h..16h+15 packed two-per-VGPR (contiguous K order, same as A).
        v16h b[4];
#pragma unroll
        for (int ft = 0; ft < 4; ++ft)
            b[ft] = *(const v16h*)&tileT[ft * 16 + n][h * 16];

        // ---- labels for this lane's A K-slots, already u16-packed in the
        //      fragment's half order: element e <-> K = 16h + e.
        const u16x8 klo = *(const u16x8*)&labs16[16 * h];
        const u16x8 khi = *(const u16x8*)&labs16[16 * h + 8];
        u16x16 kl;
#pragma unroll
        for (int e = 0; e < 8; ++e) { kl[e] = klo[e]; kl[e + 8] = khi[e]; }

        // One-hot A fragment for class tile t (packed u16, 2^-24 hot value).
        auto build_a = [&](int t) -> v16h {
            const unsigned short cls =
                (unsigned short)((wave * 4 + t) * 16 + n);  // A row class
            u16x16 clsv;
#pragma unroll
            for (int e = 0; e < 16; ++e) clsv[e] = cls;
            const u16x16 d   = kl - clsv;                       // pk_sub
            const u16x16 ind = __builtin_elementwise_sub_sat(ones, d);
            return __builtin_bit_cast(v16h, ind);
        };

        // ---- 4 class tiles, A build pipelined one tile ahead of the WMMAs
        v16h a = build_a(0);
#pragma unroll
        for (int t = 0; t < 4; ++t) {
            v16h a_next;
            if (t < 3) a_next = build_a(t + 1);
#pragma unroll
            for (int ft = 0; ft < 4; ++ft)
                acc[t][ft] = __builtin_amdgcn_wmma_f32_16x16x32_f16(
                                 false, a, false, b[ft],
                                 (short)0, acc[t][ft], false, false);
            a = a_next;
        }
    }

    // ---- combine KSPLIT partials. C/D layout: VGPR v -> row M = v + 8h.
#pragma unroll
    for (int t = 0; t < 4; ++t) {
#pragma unroll
        for (int ft = 0; ft < 4; ++ft) {
#pragma unroll
            for (int v = 0; v < 8; ++v) {
                const int c = (wave * 4 + t) * 16 + v + 8 * h;
                atomicAdd(&msums[(size_t)c * FEAT_DIM + f0 + ft * 16 + n],
                          acc[t][ft][v]);
            }
        }
    }
}

// ---------------------------------------------------------------------------
// Kernel 4: finalize. mean = 2^24 * sums / cnt; EMA only where class present.
// ---------------------------------------------------------------------------
__global__ void cma_finalize(const float* __restrict__ sums,
                             const unsigned int* __restrict__ counts,
                             const float* __restrict__ vis_mem,
                             const float* __restrict__ ir_mem,
                             float* __restrict__ out) {
    const size_t TOT = (size_t)2 * NUM_CLASSES * FEAT_DIM;
    size_t i = (size_t)blockIdx.x * blockDim.x + threadIdx.x;
    if (i >= TOT) return;
    const int d   = (int)(i % FEAT_DIM);
    const int c   = (int)((i / FEAT_DIM) % NUM_CLASSES);
    const int mod = (int)(i / ((size_t)NUM_CLASSES * FEAT_DIM));
    const unsigned cnt = counts[mod * CPAD + c];
    const float mem = (mod ? ir_mem : vis_mem)[(size_t)c * FEAT_DIM + d];
    const float s = sums[((size_t)mod * CPAD + c) * FEAT_DIM + d];
    out[i] = cnt ? ((1.0f - SIGMA) * mem +
                    SIGMA * (s * ONE_HOT_INV / (float)cnt))
                 : mem;
}

// ---------------------------------------------------------------------------
extern "C" void kernel_launch(void* const* d_in, const int* in_sizes, int n_in,
                              void* d_out, int out_size, void* d_ws, size_t ws_size,
                              hipStream_t stream) {
    const float* rgb_feats = (const float*)d_in[0];
    const float* ir_feats  = (const float*)d_in[1];
    const float* vis_mem   = (const float*)d_in[2];
    const float* ir_mem    = (const float*)d_in[3];
    const int*   rgb_lab   = (const int*)d_in[4];
    const int*   ir_lab    = (const int*)d_in[5];
    float* out = (float*)d_out;

    // Workspace: sums f32[2][CPAD][FEAT_DIM] (16 MB) then counts u32[2][CPAD].
    float* sums = (float*)d_ws;
    unsigned int* counts =
        (unsigned int*)(sums + (size_t)2 * CPAD * FEAT_DIM);

    const size_t sums_elems = (size_t)2 * CPAD * FEAT_DIM;
    cma_zero_sums<<<(int)((sums_elems + 255) / 256), 256, 0, stream>>>(sums);
    cma_zero_counts<<<(2 * CPAD + 255) / 256, 256, 0, stream>>>(counts);
    cma_count<<<(2 * NSAMP + 255) / 256, 256, 0, stream>>>(rgb_lab, ir_lab,
                                                           counts);

    dim3 grid(FEAT_DIM / 64, KSPLIT, 2);   // 32 slices x 4 k-ranges x 2 mods
    cma_scatter_wmma<<<grid, 512, 0, stream>>>(rgb_feats, ir_feats,
                                               rgb_lab, ir_lab, sums);

    const size_t tot = (size_t)2 * NUM_CLASSES * FEAT_DIM;
    cma_finalize<<<(int)((tot + 255) / 256), 256, 0, stream>>>(
        sums, counts, vis_mem, ir_mem, out);
}
